// EdgeUpdateLayerNodeAttention_42425686950358
// MI455X (gfx1250) — matrix operand order repaired and checked by reading, not compiled, not run on hardware
//
#include <hip/hip_runtime.h>
#include <hip/hip_bf16.h>

// ---------------------------------------------------------------------------
// Problem constants (from reference)
// ---------------------------------------------------------------------------
#define E_TOTAL 262144
#define EDIM 256
#define NDIM 256
#define HEADS 8
#define DH 32

// Tiling
#define MTILE 64          // edges per block
#define LDH 264           // bf16 halves per LDS row (256 + 8 pad), 528B = 33*16
#define LDF 264           // floats per LDS row for f32 buffers
#define LDT 40            // halves per row of FFN chunk buffer (32 + 8 pad)
#define TBYTES (MTILE * LDH * 2)   // 33792 bytes per bf16 tile

typedef __attribute__((ext_vector_type(16))) __bf16 v16bf;
typedef __attribute__((ext_vector_type(8)))  __bf16 v8bf;
typedef __attribute__((ext_vector_type(4)))  __bf16 v4bf;
typedef __attribute__((ext_vector_type(8)))  float  v8f;

// ---------------------------------------------------------------------------
// bf16 <-> f32 helpers (RNE) and packed-pair helpers
// ---------------------------------------------------------------------------
static __device__ __forceinline__ __bf16 f2bf(float f) {
  unsigned u = __builtin_bit_cast(unsigned, f);
  unsigned r = u + 0x7fffu + ((u >> 16) & 1u);
  unsigned short h = (unsigned short)(r >> 16);
  return __builtin_bit_cast(__bf16, h);
}
// Unpack one dword holding two bf16 into two floats (lo, hi).
static __device__ __forceinline__ void unp2(unsigned u, float& lo, float& hi) {
  hi = __builtin_bit_cast(float, u & 0xffff0000u);
  lo = __builtin_bit_cast(float, u << 16);
}
// Pack two floats into one dword of two bf16 (RNE each).
static __device__ __forceinline__ unsigned pk2(float lo, float hi) {
  unsigned a = __builtin_bit_cast(unsigned, lo);
  unsigned b = __builtin_bit_cast(unsigned, hi);
  unsigned ra = (a + 0x7fffu + ((a >> 16) & 1u)) >> 16;
  unsigned rb = (b + 0x7fffu + ((b >> 16) & 1u)) & 0xffff0000u;
  return rb | ra;
}

// ---------------------------------------------------------------------------
// WMMA fragment loaders
// A fragment (16x32 bf16) from LDS, row-major with row stride ldw halves.
// ISA layout: lane l -> row = l&15; lanes 0-15 take K chunks {0..7,16..23},
// lanes 16-31 take {8..15,24..31}. Two 16B ds_load_b128 per lane.
// ---------------------------------------------------------------------------
union AFrag { v16bf v; struct { v8bf lo, hi; } h; };

static __device__ __forceinline__ v16bf ldsA(const __bf16* sm, int ldw,
                                             int mbase, int k0, int lane) {
  AFrag a;
  const __bf16* p = sm + (size_t)(mbase + (lane & 15)) * ldw + k0 + ((lane >> 4) << 3);
  a.h.lo = *(const v8bf*)p;
  a.h.hi = *(const v8bf*)(p + 16);
  return a.v;
}

// B fragment (32x16 bf16) from pre-packed global weights: for block (kt,nt),
// lane l's 16 halves stored contiguously -> one 32B load (L2-resident).
static __device__ __forceinline__ v16bf ldgB(const __bf16* Wp, int ntiles,
                                             int kt, int nt, int lane) {
  return *(const v16bf*)(Wp + ((((size_t)kt * ntiles + nt) * 32 + lane) << 4));
}

static __device__ __forceinline__ v8f wmma_bf16(v16bf a, v16bf b, v8f c) {
  return __builtin_amdgcn_wmma_f32_16x16x32_bf16(false, a, false, b, (short)0, c,
                                                 false, false);
}

// ---------------------------------------------------------------------------
// Full [64,256] = A[64,256] x W[256,256] GEMM, result to bf16 LDS (+bias).
// 8 waves: wave w -> rows (w&3)*16..+15, cols (w>>2)*128..+127 (8 C tiles).
// ---------------------------------------------------------------------------
static __device__ __forceinline__ void gemm64x256_bf16(const __bf16* Asm,
                                                       const __bf16* Wp,
                                                       const float* __restrict__ bias,
                                                       __bf16* Dsm,
                                                       int wave, int lane) {
  const int mb  = (wave & 3) << 4;
  const int nt0 = (wave >> 2) << 3;
  v8f acc[8] = {};
#pragma unroll
  for (int kt = 0; kt < 8; ++kt) {
    v16bf a = ldsA(Asm, LDH, mb, kt << 5, lane);
#pragma unroll
    for (int j = 0; j < 8; ++j) {
      v16bf b = ldgB(Wp, 16, kt, nt0 + j, lane);
      acc[j] = wmma_bf16(a, b, acc[j]);
    }
  }
  const int r0 = mb + ((lane >> 4) << 3);
#pragma unroll
  for (int j = 0; j < 8; ++j) {
    const int col = ((nt0 + j) << 4) + (lane & 15);
    const float bv = bias[col];
#pragma unroll
    for (int i = 0; i < 8; ++i)
      Dsm[(size_t)(r0 + i) * LDH + col] = f2bf(acc[j][i] + bv);
  }
}

// Same GEMM but D = A x Wo + bo + x_residual(global), stored f32 to LDS.
static __device__ __forceinline__ void gemm64x256_resid(const __bf16* Asm,
                                                        const __bf16* Wp,
                                                        const float* __restrict__ bias,
                                                        const float* __restrict__ xg,
                                                        float* U,
                                                        int wave, int lane) {
  const int mb  = (wave & 3) << 4;
  const int nt0 = (wave >> 2) << 3;
  v8f acc[8] = {};
#pragma unroll
  for (int kt = 0; kt < 8; ++kt) {
    v16bf a = ldsA(Asm, LDH, mb, kt << 5, lane);
#pragma unroll
    for (int j = 0; j < 8; ++j) {
      v16bf b = ldgB(Wp, 16, kt, nt0 + j, lane);
      acc[j] = wmma_bf16(a, b, acc[j]);
    }
  }
  const int r0 = mb + ((lane >> 4) << 3);
#pragma unroll
  for (int j = 0; j < 8; ++j) {
    const int col = ((nt0 + j) << 4) + (lane & 15);
    const float bv = bias[col];
#pragma unroll
    for (int i = 0; i < 8; ++i)
      U[(size_t)(r0 + i) * LDF + col] =
          acc[j][i] + bv + xg[(size_t)(r0 + i) * EDIM + col];
  }
}

// ---------------------------------------------------------------------------
// Weight pre-pack: f32 [K,N] row-major -> bf16 WMMA-B-fragment blocks.
// One thread per (kt, nt, lane); writes 16 contiguous halves.
// ---------------------------------------------------------------------------
__global__ void pack_weights(const float* __restrict__ W, __bf16* __restrict__ Wp,
                             int K, int N) {
  const int ntiles = N >> 4;
  const int total  = (K >> 5) * ntiles * 32;
  int idx = blockIdx.x * blockDim.x + threadIdx.x;
  if (idx >= total) return;
  const int lane = idx & 31;
  const int t    = idx >> 5;
  const int nt   = t % ntiles;
  const int kt   = t / ntiles;
  const int n    = lane & 15;
  const int kh   = lane >> 4;
  __bf16* dst = Wp + ((size_t)idx << 4);
#pragma unroll
  for (int j = 0; j < 16; ++j) {
    float v = W[(size_t)((kt << 5) + (kh << 3) + (j & 7) + ((j >> 3) << 4)) * N + (nt << 4) + n];
    dst[j] = f2bf(v);
  }
}

// ---------------------------------------------------------------------------
// Fused edge-update layer kernel
// ---------------------------------------------------------------------------
__global__ __launch_bounds__(256)
void edge_layer_kernel(const float* __restrict__ x,
                       const float* __restrict__ srcx,
                       const float* __restrict__ dstx,
                       const __bf16* __restrict__ Wqp, const __bf16* __restrict__ Wkp,
                       const __bf16* __restrict__ Wvp, const __bf16* __restrict__ Wop,
                       const __bf16* __restrict__ W1p, const __bf16* __restrict__ W2p,
                       const float* __restrict__ bq, const float* __restrict__ bk,
                       const float* __restrict__ bv, const float* __restrict__ bo,
                       const float* __restrict__ b1, const float* __restrict__ b2,
                       const float* __restrict__ g1, const float* __restrict__ be1,
                       const float* __restrict__ g2, const float* __restrict__ be2,
                       float* __restrict__ out) {
  extern __shared__ char smem[];
  __bf16* xs  = (__bf16*)(smem + 0 * TBYTES);
  __bf16* ssm = (__bf16*)(smem + 1 * TBYTES);
  __bf16* dsm = (__bf16*)(smem + 2 * TBYTES);
  __bf16* qs  = (__bf16*)(smem + 3 * TBYTES);
  __bf16* ks  = (__bf16*)(smem + 4 * TBYTES);
  __bf16* kd  = (__bf16*)(smem + 5 * TBYTES);
  __bf16* vs  = (__bf16*)(smem + 6 * TBYTES);
  __bf16* vd  = (__bf16*)(smem + 7 * TBYTES);
  float*  red = (float*)(smem + 8 * TBYTES);
  // Region overlays (lifetimes are disjoint, see phase ordering):
  __bf16* ctx = xs;                 // attention output  (xs dead after Q GEMM)
  float*  uf  = (float*)ssm;        // f32 [64][LDF] spans ssm+dsm (inputs dead)
  float*  hf  = (float*)ks;         // f32 [64][LDF] spans ks+kd  (keys dead)
  __bf16* hb  = vs;                 // bf16 h         (vs dead)
  __bf16* tb  = vd;                 // FFN chunk buf  (vd dead)
  float* rs = red;                  // [64][4] partial sums
  float* rq = red + 256;            // [64][4] partial sumsq

  const int tid  = threadIdx.x;
  const int lane = tid & 31;
  const int wave = tid >> 5;
  const long e0  = (long)blockIdx.x * MTILE;
  const float* xg = x + (size_t)e0 * EDIM;

  // ---- Phase 0: load x/src/dst tiles, convert to bf16 in LDS -------------
  {
    const float4* xg4 = (const float4*)(x    + (size_t)e0 * EDIM);
    const float4* sg4 = (const float4*)(srcx + (size_t)e0 * NDIM);
    const float4* dg4 = (const float4*)(dstx + (size_t)e0 * NDIM);
    for (int i = tid; i < MTILE * 64; i += 256) {
      const int r  = i >> 6;
      const int q4 = i & 63;
      const int c  = q4 << 2;
      float4 a = xg4[r * 64 + q4];
      float4 b = sg4[r * 64 + q4];
      float4 d = dg4[r * 64 + q4];
      v4bf pa, pb, pd;
      pa[0] = f2bf(a.x); pa[1] = f2bf(a.y); pa[2] = f2bf(a.z); pa[3] = f2bf(a.w);
      pb[0] = f2bf(b.x); pb[1] = f2bf(b.y); pb[2] = f2bf(b.z); pb[3] = f2bf(b.w);
      pd[0] = f2bf(d.x); pd[1] = f2bf(d.y); pd[2] = f2bf(d.z); pd[3] = f2bf(d.w);
      *(v4bf*)(xs  + (size_t)r * LDH + c) = pa;
      *(v4bf*)(ssm + (size_t)r * LDH + c) = pb;
      *(v4bf*)(dsm + (size_t)r * LDH + c) = pd;
    }
  }
  __syncthreads();

  // Warm caches for the x residual tile consumed in phase 3 (hidden behind
  // the five QKV GEMMs). Emits global_prefetch_b8; no counter impact.
  if (wave == 0) {
    for (int rr = lane; rr < MTILE; rr += 32)
      __builtin_prefetch(xg + (size_t)rr * EDIM, 0, 0);
  }

  // ---- Phase 1: Q/K/V projections (5 WMMA GEMMs) -------------------------
  gemm64x256_bf16(xs,  Wqp, bq, qs, wave, lane);
  gemm64x256_bf16(ssm, Wkp, bk, ks, wave, lane);
  gemm64x256_bf16(dsm, Wkp, bk, kd, wave, lane);
  gemm64x256_bf16(ssm, Wvp, bv, vs, wave, lane);
  gemm64x256_bf16(dsm, Wvp, bv, vd, wave, lane);
  __syncthreads();

  // ---- Phase 2: 2-key softmax cross-attention (vectorized VALU) ----------
  // 16B (ds_load_b128) packed-bf16 loads; pair unpack is 1 shift + 1 mask.
  {
    const float scale = 0.17677669529663687f;  // 1/sqrt(32)
    for (int t = tid; t < MTILE * HEADS; t += 256) {
      const int r  = t >> 3;
      const int h  = t & 7;
      const size_t off = (size_t)r * LDH + h * DH;  // 16B-aligned
      const uint4* qp  = (const uint4*)(qs + off);
      const uint4* k1p = (const uint4*)(ks + off);
      const uint4* k2p = (const uint4*)(kd + off);
      float s0 = 0.f, s1 = 0.f;
#pragma unroll
      for (int g = 0; g < 4; ++g) {
        uint4 qu = qp[g], u1 = k1p[g], u2 = k2p[g];
        const unsigned qa[4] = {qu.x, qu.y, qu.z, qu.w};
        const unsigned ka[4] = {u1.x, u1.y, u1.z, u1.w};
        const unsigned kb[4] = {u2.x, u2.y, u2.z, u2.w};
#pragma unroll
        for (int w = 0; w < 4; ++w) {
          float ql, qh, xl, xh, yl, yh;
          unp2(qa[w], ql, qh);
          unp2(ka[w], xl, xh);
          unp2(kb[w], yl, yh);
          s0 += ql * xl + qh * xh;
          s1 += ql * yl + qh * yh;
        }
      }
      s0 *= scale; s1 *= scale;
      const float m   = fmaxf(s0, s1);
      const float x0  = __expf(s0 - m);
      const float x1  = __expf(s1 - m);
      const float rcp = 1.f / (x0 + x1);
      const float a0  = x0 * rcp, a1 = x1 * rcp;
      const uint4* v1p = (const uint4*)(vs + off);
      const uint4* v2p = (const uint4*)(vd + off);
      uint4* cp = (uint4*)(ctx + off);
#pragma unroll
      for (int g = 0; g < 4; ++g) {
        uint4 u1 = v1p[g], u2 = v2p[g], o;
        const unsigned va[4] = {u1.x, u1.y, u1.z, u1.w};
        const unsigned vb[4] = {u2.x, u2.y, u2.z, u2.w};
        unsigned ow[4];
#pragma unroll
        for (int w = 0; w < 4; ++w) {
          float xl, xh, yl, yh;
          unp2(va[w], xl, xh);
          unp2(vb[w], yl, yh);
          ow[w] = pk2(a0 * xl + a1 * yl, a0 * xh + a1 * yh);
        }
        o.x = ow[0]; o.y = ow[1]; o.z = ow[2]; o.w = ow[3];
        cp[g] = o;
      }
    }
  }
  __syncthreads();

  // ---- Phase 3: out = ctx @ Wo + bo + x  (f32 to LDS) --------------------
  gemm64x256_resid(ctx, Wop, bo, xg, uf, wave, lane);
  __syncthreads();

  // ---- LayerNorm 1: h = LN(uf); keep f32 (hf) and bf16 (hb) copies -------
  {
    const int r = tid >> 2, part = tid & 3;
    const float4* rowp4 = (const float4*)(uf + (size_t)r * LDF) + (part << 4);
    float s = 0.f, q = 0.f;
#pragma unroll
    for (int i = 0; i < 16; ++i) {
      float4 v = rowp4[i];
      s += v.x + v.y + v.z + v.w;
      q += v.x * v.x + v.y * v.y + v.z * v.z + v.w * v.w;
    }
    rs[(r << 2) + part] = s;
    rq[(r << 2) + part] = q;
    __syncthreads();
    const float mn = (rs[r * 4] + rs[r * 4 + 1] + rs[r * 4 + 2] + rs[r * 4 + 3]) *
                     (1.f / 256.f);
    const float ms = (rq[r * 4] + rq[r * 4 + 1] + rq[r * 4 + 2] + rq[r * 4 + 3]) *
                     (1.f / 256.f);
    const float iv = rsqrtf(ms - mn * mn + 1e-5f);
    const float4* gp = (const float4*)g1  + (part << 4);
    const float4* bp = (const float4*)be1 + (part << 4);
    float4* hf4 = (float4*)(hf + (size_t)r * LDF) + (part << 4);
    __bf16* hbr = hb + (size_t)r * LDH + (part << 6);
#pragma unroll
    for (int i = 0; i < 16; ++i) {
      float4 v = rowp4[i], g = gp[i], b = bp[i];
      float4 o;
      o.x = (v.x - mn) * iv * g.x + b.x;
      o.y = (v.y - mn) * iv * g.y + b.y;
      o.z = (v.z - mn) * iv * g.z + b.z;
      o.w = (v.w - mn) * iv * g.w + b.w;
      hf4[i] = o;
      v4bf p;
      p[0] = f2bf(o.x); p[1] = f2bf(o.y); p[2] = f2bf(o.z); p[3] = f2bf(o.w);
      *(v4bf*)(hbr + (i << 2)) = p;
    }
  }
  __syncthreads();

  // ---- Phase 4: FFN streamed in 32-wide hidden chunks --------------------
  // ff2 += relu(h @ W1[:,c*32:+32] + b1) @ W2[c*32:+32,:], 32 chunks; the
  // 1024-wide hidden activation never leaves LDS.
  {
    const int mb   = (wave & 3) << 4;
    const int nsel = wave >> 2;  // 0 or 1 -> which 128-col half / 16-col subtile
    v8f ff[8] = {};
    for (int c = 0; c < 32; ++c) {
      // GEMM1: this wave computes the (mb, nsel*16) tile of [64,32]
      v8f t1 = {};
#pragma unroll
      for (int kt = 0; kt < 8; ++kt) {
        v16bf a = ldsA(hb, LDH, mb, kt << 5, lane);
        v16bf b = ldgB(W1p, 64, kt, (c << 1) + nsel, lane);
        t1 = wmma_bf16(a, b, t1);
      }
      {
        const int col = (nsel << 4) + (lane & 15);
        const float bvv = b1[(c << 5) + col];
        const int r0 = mb + ((lane >> 4) << 3);
#pragma unroll
        for (int i = 0; i < 8; ++i) {
          float v = t1[i] + bvv;
          tb[(size_t)(r0 + i) * LDT + col] = f2bf(v > 0.f ? v : 0.f);
        }
      }
      __syncthreads();
      // GEMM2: single K=32 step accumulating into this wave's [16,128] stripe
      v16bf a2 = ldsA(tb, LDT, mb, 0, lane);
#pragma unroll
      for (int j = 0; j < 8; ++j) {
        v16bf b = ldgB(W2p, 16, c, (nsel << 3) + j, lane);
        ff[j] = wmma_bf16(a2, b, ff[j]);
      }
      __syncthreads();
    }
    // u2 = h + ff + b2 -> uf
    const int r0 = mb + ((lane >> 4) << 3);
#pragma unroll
    for (int j = 0; j < 8; ++j) {
      const int col = (nsel << 7) + (j << 4) + (lane & 15);
      const float bvv = b2[col];
#pragma unroll
      for (int i = 0; i < 8; ++i)
        uf[(size_t)(r0 + i) * LDF + col] =
            ff[j][i] + bvv + hf[(size_t)(r0 + i) * LDF + col];
    }
  }
  __syncthreads();

  // ---- LayerNorm 2 + vectorized store to global --------------------------
  {
    const int r = tid >> 2, part = tid & 3;
    const float4* rowp4 = (const float4*)(uf + (size_t)r * LDF) + (part << 4);
    float s = 0.f, q = 0.f;
#pragma unroll
    for (int i = 0; i < 16; ++i) {
      float4 v = rowp4[i];
      s += v.x + v.y + v.z + v.w;
      q += v.x * v.x + v.y * v.y + v.z * v.z + v.w * v.w;
    }
    rs[(r << 2) + part] = s;
    rq[(r << 2) + part] = q;
    __syncthreads();
    const float mn = (rs[r * 4] + rs[r * 4 + 1] + rs[r * 4 + 2] + rs[r * 4 + 3]) *
                     (1.f / 256.f);
    const float ms = (rq[r * 4] + rq[r * 4 + 1] + rq[r * 4 + 2] + rq[r * 4 + 3]) *
                     (1.f / 256.f);
    const float iv = rsqrtf(ms - mn * mn + 1e-5f);
    const float4* gp = (const float4*)g2  + (part << 4);
    const float4* bp = (const float4*)be2 + (part << 4);
    float4* og4 = (float4*)(out + (size_t)(e0 + r) * EDIM) + (part << 4);
#pragma unroll
    for (int i = 0; i < 16; ++i) {
      float4 v = rowp4[i], g = gp[i], b = bp[i];
      float4 o;
      o.x = (v.x - mn) * iv * g.x + b.x;
      o.y = (v.y - mn) * iv * g.y + b.y;
      o.z = (v.z - mn) * iv * g.z + b.z;
      o.w = (v.w - mn) * iv * g.w + b.w;
      og4[i] = o;
    }
  }
}

// ---------------------------------------------------------------------------
// Launcher
// ---------------------------------------------------------------------------
extern "C" void kernel_launch(void* const* d_in, const int* in_sizes, int n_in,
                              void* d_out, int out_size, void* d_ws, size_t ws_size,
                              hipStream_t stream) {
  const float* x    = (const float*)d_in[0];
  const float* srcx = (const float*)d_in[1];
  const float* dstx = (const float*)d_in[2];
  const float* Wq = (const float*)d_in[3];  const float* bq = (const float*)d_in[4];
  const float* Wk = (const float*)d_in[5];  const float* bk = (const float*)d_in[6];
  const float* Wv = (const float*)d_in[7];  const float* bv = (const float*)d_in[8];
  const float* Wo = (const float*)d_in[9];  const float* bo = (const float*)d_in[10];
  const float* W1 = (const float*)d_in[11]; const float* b1 = (const float*)d_in[12];
  const float* W2 = (const float*)d_in[13]; const float* b2 = (const float*)d_in[14];
  const float* g1 = (const float*)d_in[15]; const float* be1 = (const float*)d_in[16];
  const float* g2 = (const float*)d_in[17]; const float* be2 = (const float*)d_in[18];
  float* out = (float*)d_out;

  char* ws = (char*)d_ws;
  __bf16* Wqp = (__bf16*)(ws + 0);
  __bf16* Wkp = (__bf16*)(ws + 131072);
  __bf16* Wvp = (__bf16*)(ws + 262144);
  __bf16* Wop = (__bf16*)(ws + 393216);
  __bf16* W1p = (__bf16*)(ws + 524288);
  __bf16* W2p = (__bf16*)(ws + 1048576);

  // Pack weights to bf16 fragment-friendly layout (tiny, L2-resident).
  pack_weights<<<16, 256, 0, stream>>>(Wq, Wqp, 256, 256);
  pack_weights<<<16, 256, 0, stream>>>(Wk, Wkp, 256, 256);
  pack_weights<<<16, 256, 0, stream>>>(Wv, Wvp, 256, 256);
  pack_weights<<<16, 256, 0, stream>>>(Wo, Wop, 256, 256);
  pack_weights<<<64, 256, 0, stream>>>(W1, W1p, 256, 1024);
  pack_weights<<<64, 256, 0, stream>>>(W2, W2p, 1024, 256);

  const size_t smem = (size_t)8 * TBYTES + 4096;  // 274432 B (<320 KB WGP LDS)
  edge_layer_kernel<<<E_TOTAL / MTILE, 256, smem, stream>>>(
      x, srcx, dstx, Wqp, Wkp, Wvp, Wop, W1p, W2p,
      bq, bk, bv, bo, b1, b2, g1, be1, g2, be2, out);
}
